// InformerModel_25761213841479
// MI455X (gfx1250) — matrix operand order repaired and checked by reading, not compiled, not run on hardware
//
#include <hip/hip_runtime.h>
#include <math.h>

// ---------------------------------------------------------------------------
// Informer forward for MI455X (gfx1250, wave32).
// All big GEMMs (QKV/O projections, FFNs, distill conv) run on
// v_wmma_f32_16x16x32_f16 (f16 inputs staged via LDS, f32 accumulate),
// double-buffered so next-tile global loads overlap with WMMA compute.
// LDS commits use packed f16x2 ds_store_b32.
// Attention scoring / top-k / softmax pieces are tiny (<1% FLOPs) -> VALU.
// ---------------------------------------------------------------------------

#define DMODEL 512
#define NHEAD  8
#define DHEAD  64
#define DFFN   2048
#define NB     4

typedef __attribute__((ext_vector_type(16))) _Float16 v16h;
typedef __attribute__((ext_vector_type(8)))  float    v8f;

union F16x16 { v16h v; unsigned u[8]; };
union H2     { _Float16 h[2]; unsigned u; };

// ---------------------------------------------------------------------------
// Generic WMMA GEMM:  Cout[M,N] = op( A' * B' ) with
//   A'(row r, col k) = A[(row mapped via aseg/ashift)*lda + k]   (fp32)
//   B'(k, n)         = Bm[bOff + k*ldbK + n*ldbN]                (fp32)
// flags: bit0 += Cin, bit1 += bias[n], bit2 gelu(exact erf)
// M, N multiples of 128; K multiple of 32 (true for all calls here).
// Double-buffered LDS; staging is organized in k-pairs so A global loads
// merge to b64 and LDS commits are single packed ds_store_b32 per pair.
// ---------------------------------------------------------------------------
__global__ __launch_bounds__(256) void k_gemm(
    const float* __restrict__ A, const float* __restrict__ Bm,
    const float* __restrict__ bias, const float* __restrict__ Cin,
    float* __restrict__ Cout,
    int M, int N, int K, int lda,
    long long bOff, int ldbK, int ldbN,
    int ashift, int aseg, int flags)
{
    (void)M;
    __shared__ _Float16 As[2][128][32];   // [buf][m][k]
    __shared__ _Float16 Bs[2][128][32];   // [buf][n][k]  (column-major B tile)

    const int tid    = threadIdx.x;
    const int wave   = tid >> 5;
    const int lane   = tid & 31;
    const int laneLo = lane & 15;
    const int laneHi = lane >> 4;
    const int waveM  = wave & 3;       // 4 waves along M (32 rows each)
    const int waveN  = wave >> 2;      // 2 waves along N (64 cols each)
    const int rowBase = blockIdx.y * 128;
    const int colBase = blockIdx.x * 128;
    const int ntiles  = K >> 5;

    v8f acc[2][4] = {};
    float ra[16], rb[16];

    // ---- load one 128x32 tile pair (global -> regs), k-pair decomposition --
    // pair pp = tid + 256*jp : m = pp>>4 , k2 = (pp&15)*2  (+ kk)
    auto loadTile = [&](int kk) {
        #pragma unroll
        for (int jp = 0; jp < 8; ++jp) {
            int pp = tid + 256 * jp;
            int m  = pp >> 4;
            int k2 = ((pp & 15) << 1) + kk;
            int gm = rowBase + m;
            float a0, a1;
            if (aseg > 0) {
                int bb = gm / aseg;
                int l  = gm - bb * aseg + ashift;
                if (l >= 0 && l < aseg) {
                    size_t base = (size_t)(bb * aseg + l) * lda + k2;
                    a0 = A[base]; a1 = A[base + 1];
                } else {
                    a0 = 0.0f; a1 = 0.0f;
                }
            } else {
                size_t base = (size_t)gm * lda + k2;
                a0 = A[base]; a1 = A[base + 1];
            }
            ra[2 * jp] = a0; ra[2 * jp + 1] = a1;
            size_t bb0 = bOff + (size_t)k2 * ldbK + (size_t)(colBase + m) * ldbN;
            rb[2 * jp] = Bm[bb0]; rb[2 * jp + 1] = Bm[bb0 + ldbK];
        }
    };
    // ---- commit regs -> LDS buffer `buf` as packed f16x2 b32 stores --------
    auto commitTile = [&](int buf) {
        #pragma unroll
        for (int jp = 0; jp < 8; ++jp) {
            int pp = tid + 256 * jp;
            int m  = pp >> 4;
            int k2 = (pp & 15) << 1;
            H2 ha; ha.h[0] = (_Float16)ra[2 * jp]; ha.h[1] = (_Float16)ra[2 * jp + 1];
            H2 hb; hb.h[0] = (_Float16)rb[2 * jp]; hb.h[1] = (_Float16)rb[2 * jp + 1];
            *(unsigned*)&As[buf][m][k2] = ha.u;
            *(unsigned*)&Bs[buf][m][k2] = hb.u;
        }
    };

    loadTile(0);
    commitTile(0);
    __syncthreads();

    for (int t = 0; t < ntiles; ++t) {
        const int cur = t & 1;

        // ---- prefetch tile t+1 into registers (overlaps with WMMA below) --
        if (t + 1 < ntiles) loadTile((t + 1) << 5);

        // ---- compute on LDS buf `cur` ----
        v16h af[2], bf[4];
        #pragma unroll
        for (int am = 0; am < 2; ++am) {
            int m = waveM * 32 + am * 16 + laneLo;
            F16x16 tt;
            #pragma unroll
            for (int p = 0; p < 8; ++p) {
                int k0 = ((p >> 2) << 4) + (laneHi << 3) + ((p & 3) << 1);
                tt.u[p] = *(const unsigned*)&As[cur][m][k0];
            }
            af[am] = tt.v;
        }
        #pragma unroll
        for (int bn = 0; bn < 4; ++bn) {
            int n = waveN * 64 + bn * 16 + laneLo;
            F16x16 tt;
            #pragma unroll
            for (int p = 0; p < 8; ++p) {
                int k0 = ((p >> 2) << 4) + (laneHi << 3) + ((p & 3) << 1);
                tt.u[p] = *(const unsigned*)&Bs[cur][n][k0];
            }
            bf[bn] = tt.v;
        }
        #pragma unroll
        for (int am = 0; am < 2; ++am)
            #pragma unroll
            for (int bn = 0; bn < 4; ++bn)
                acc[am][bn] = __builtin_amdgcn_wmma_f32_16x16x32_f16(
                    false, af[am], false, bf[bn], (short)0, acc[am][bn],
                    false, false);

        // ---- commit prefetched tile into the other LDS buffer ----
        if (t + 1 < ntiles) {
            __syncthreads();   // everyone done reading buf (1-cur) from t-1
            commitTile(1 - cur);
            __syncthreads();
        }
    }

    // epilogue: D layout -> m = 8*laneHi + j, n = laneLo
    #pragma unroll
    for (int am = 0; am < 2; ++am)
        #pragma unroll
        for (int bn = 0; bn < 4; ++bn)
            #pragma unroll
            for (int j = 0; j < 8; ++j) {
                int row = rowBase + waveM * 32 + am * 16 + laneHi * 8 + j;
                int col = colBase + waveN * 64 + bn * 16 + laneLo;
                size_t o = (size_t)row * N + col;
                float v = acc[am][bn][j];
                if (flags & 1) v += Cin[o];
                if (flags & 2) v += bias[col];
                if (flags & 4) v = 0.5f * v * (1.0f + erff(v * 0.70710678118f));
                Cout[o] = v;
            }
}

// ---------------------------------------------------------------------------
// embed: out[(b*L+l)*512+d] = x[b*L+l]*w[d] + b[d] + PE(l,d)
// ---------------------------------------------------------------------------
__global__ void k_embed(const float* __restrict__ x, const float* __restrict__ w,
                        const float* __restrict__ b, float* __restrict__ out,
                        int L, size_t total)
{
    size_t i = (size_t)blockIdx.x * 256 + threadIdx.x;
    if (i >= total) return;
    int d = (int)(i % DMODEL);
    size_t r = i / DMODEL;
    int l = (int)(r % L);
    int pair = d >> 1;
    float div = expf(-9.210340371976184f * (float)(2 * pair) / (float)DMODEL);
    float ang = (float)l * div;
    float pe  = (d & 1) ? cosf(ang) : sinf(ang);
    out[i] = x[r] * w[d] + b[d] + pe;
}

// ---------------------------------------------------------------------------
// sampled QK scores: M[b,h,l] = max_s(QK) - mean_s(QK)   (wave per (b,h,l))
// ---------------------------------------------------------------------------
__global__ void k_sample(const float* __restrict__ Q, const float* __restrict__ K,
                         const int* __restrict__ idx, float* __restrict__ Mout,
                         int L, int u)
{
    int gid  = blockIdx.x * 8 + (threadIdx.x >> 5);   // over B*H*L
    int lane = threadIdx.x & 31;
    int l  = gid % L;
    int bh = gid / L;
    int h  = bh & 7;
    int b  = bh >> 3;
    size_t qb = ((size_t)(b * L + l) * DMODEL) + h * DHEAD;
    float q0 = Q[qb + lane], q1 = Q[qb + 32 + lane];
    float mx = -3.4e38f, sm = 0.0f;
    for (int s = 0; s < u; ++s) {
        int kr = idx[l * u + s];
        size_t kb = ((size_t)(b * L + kr) * DMODEL) + h * DHEAD;
        float p = q0 * K[kb + lane] + q1 * K[kb + 32 + lane];
        for (int o = 16; o > 0; o >>= 1) p += __shfl_xor(p, o, 32);
        mx = fmaxf(mx, p);
        sm += p;
    }
    if (lane == 0) Mout[(size_t)bh * L + l] = mx - sm / (float)u;
}

// ---------------------------------------------------------------------------
// iterative top-u argmax per (b,h)
// ---------------------------------------------------------------------------
__global__ void k_topk(const float* __restrict__ Mv, int* __restrict__ Mtop,
                       int L, int u)
{
    int bh = blockIdx.x;
    int tid = threadIdx.x;
    __shared__ float vals[2048];
    __shared__ float rv[256];
    __shared__ int   ri[256];
    for (int i = tid; i < L; i += 256) vals[i] = Mv[(size_t)bh * L + i];
    __syncthreads();
    for (int t = 0; t < u; ++t) {
        float bv = -3.4e38f; int bi = 0;
        for (int i = tid; i < L; i += 256) {
            float v = vals[i];
            if (v > bv) { bv = v; bi = i; }
        }
        rv[tid] = bv; ri[tid] = bi;
        __syncthreads();
        for (int s = 128; s > 0; s >>= 1) {
            if (tid < s) {
                if (rv[tid + s] > rv[tid] ||
                    (rv[tid + s] == rv[tid] && ri[tid + s] < ri[tid])) {
                    rv[tid] = rv[tid + s]; ri[tid] = ri[tid + s];
                }
            }
            __syncthreads();
        }
        if (tid == 0) { Mtop[bh * u + t] = ri[0]; vals[ri[0]] = -3.4e38f; }
        __syncthreads();
    }
}

// ---------------------------------------------------------------------------
// mean of V over keys, per (b,h,e)
// ---------------------------------------------------------------------------
__global__ void k_vmean(const float* __restrict__ V, float* __restrict__ out, int L)
{
    int bh = blockIdx.x;
    int b = bh >> 3, h = bh & 7;
    int e = threadIdx.x;   // 64 threads
    float s = 0.0f;
    for (int l = 0; l < L; ++l)
        s += V[((size_t)(b * L + l) * DMODEL) + h * DHEAD + e];
    out[bh * DHEAD + e] = s / (float)L;
}

__global__ void k_fill(float* __restrict__ ctx, const float* __restrict__ vmean,
                       int L, size_t total)
{
    size_t i = (size_t)blockIdx.x * 256 + threadIdx.x;
    if (i >= total) return;
    int c = (int)(i % DMODEL);
    size_t r = i / DMODEL;
    int b = (int)(r / L);
    ctx[i] = vmean[(b * NHEAD + (c >> 6)) * DHEAD + (c & 63)];
}

// ---------------------------------------------------------------------------
// softmax attention for one (b,h,query): used for reduced ProbSparse queries
// (sel!=null: scatter into selected rows) and for dense cross-attention.
// ---------------------------------------------------------------------------
__global__ void k_attn(const float* __restrict__ Q, const float* __restrict__ K,
                       const float* __restrict__ V, const int* __restrict__ sel,
                       int nSel, int Lq, int Lk, float scale,
                       float* __restrict__ out)
{
    int tid = threadIdx.x;
    int blk = blockIdx.x;          // (b*H+h)*nSel + ui
    int ui = blk % nSel;
    int bh = blk / nSel;
    int h = bh & 7, b = bh >> 3;
    int qr = sel ? sel[bh * nSel + ui] : ui;

    __shared__ float qv[64];
    __shared__ float sc[2048];
    __shared__ float red[256];

    size_t qb = ((size_t)(b * Lq + qr) * DMODEL) + h * DHEAD;
    if (tid < 64) qv[tid] = Q[qb + tid];
    __syncthreads();

    for (int k = tid; k < Lk; k += 256) {
        size_t kb = ((size_t)(b * Lk + k) * DMODEL) + h * DHEAD;
        float s = 0.0f;
        for (int e = 0; e < 64; ++e) s += qv[e] * K[kb + e];
        sc[k] = s * scale;
    }
    __syncthreads();

    float mx = -3.4e38f;
    for (int k = tid; k < Lk; k += 256) mx = fmaxf(mx, sc[k]);
    red[tid] = mx; __syncthreads();
    for (int s = 128; s > 0; s >>= 1) {
        if (tid < s) red[tid] = fmaxf(red[tid], red[tid + s]);
        __syncthreads();
    }
    mx = red[0]; __syncthreads();

    float sm = 0.0f;
    for (int k = tid; k < Lk; k += 256) {
        float p = expf(sc[k] - mx);
        sc[k] = p;
        sm += p;
    }
    red[tid] = sm; __syncthreads();
    for (int s = 128; s > 0; s >>= 1) {
        if (tid < s) red[tid] += red[tid + s];
        __syncthreads();
    }
    sm = red[0]; __syncthreads();

    int e = tid & 63, qt = tid >> 6;
    int chunk = Lk >> 2;
    float acc = 0.0f;
    for (int k = qt * chunk; k < (qt + 1) * chunk; ++k) {
        size_t vb = ((size_t)(b * Lk + k) * DMODEL) + h * DHEAD;
        acc += sc[k] * V[vb + e];
    }
    red[qt * 64 + e] = acc;
    __syncthreads();
    if (tid < 64) {
        float r = red[tid] + red[64 + tid] + red[128 + tid] + red[192 + tid];
        out[((size_t)(b * Lq + qr) * DMODEL) + h * DHEAD + tid] = r / sm;
    }
}

// ---------------------------------------------------------------------------
// out = LayerNorm(x + r) * g + b      (block per row of 512)
// ---------------------------------------------------------------------------
__global__ void k_lnres(const float* __restrict__ x, const float* __restrict__ r,
                        const float* __restrict__ g, const float* __restrict__ b,
                        float* __restrict__ out)
{
    int row = blockIdx.x;
    int tid = threadIdx.x;     // 256 threads, 512 cols
    __shared__ float red[256];
    size_t base = (size_t)row * DMODEL;
    float v0 = x[base + tid]       + r[base + tid];
    float v1 = x[base + 256 + tid] + r[base + 256 + tid];
    red[tid] = v0 + v1; __syncthreads();
    for (int s = 128; s > 0; s >>= 1) {
        if (tid < s) red[tid] += red[tid + s];
        __syncthreads();
    }
    float mean = red[0] / 512.0f; __syncthreads();
    float d0 = v0 - mean, d1 = v1 - mean;
    red[tid] = d0 * d0 + d1 * d1; __syncthreads();
    for (int s = 128; s > 0; s >>= 1) {
        if (tid < s) red[tid] += red[tid + s];
        __syncthreads();
    }
    float inv = rsqrtf(red[0] / 512.0f + 1e-5f);
    out[base + tid]       = d0 * inv * g[tid]       + b[tid];
    out[base + 256 + tid] = d1 * inv * g[tid + 256] + b[tid + 256];
}

// ---------------------------------------------------------------------------
// distill: y = elu(batchnorm(conv_out + conv_b)); then maxpool k=3 s=2 p=1
// ---------------------------------------------------------------------------
__global__ void k_bnelu(float* __restrict__ y, const float* __restrict__ cb,
                        const float* __restrict__ g, const float* __restrict__ bb,
                        const float* __restrict__ m, const float* __restrict__ v,
                        size_t total)
{
    size_t i = (size_t)blockIdx.x * 256 + threadIdx.x;
    if (i >= total) return;
    int c = (int)(i % DMODEL);
    float x = y[i] + cb[c];
    x = (x - m[c]) * rsqrtf(v[c] + 1e-5f) * g[c] + bb[c];
    y[i] = x > 0.0f ? x : expm1f(x);
}

__global__ void k_pool(const float* __restrict__ in, float* __restrict__ out,
                       int Lin, int Lout, size_t total)
{
    size_t i = (size_t)blockIdx.x * 256 + threadIdx.x;
    if (i >= total) return;
    int c = (int)(i % DMODEL);
    size_t r = i / DMODEL;
    int lo = (int)(r % Lout);
    int b  = (int)(r / Lout);
    float mx = -3.4e38f;
    for (int dl = -1; dl <= 1; ++dl) {
        int l = 2 * lo + dl;
        if (l >= 0 && l < Lin)
            mx = fmaxf(mx, in[((size_t)(b * Lin + l)) * DMODEL + c]);
    }
    out[i] = mx;
}

// ---------------------------------------------------------------------------
// head: out[row] = dot(x[row,:512], w) + b[0]      (wave per row)
// ---------------------------------------------------------------------------
__global__ void k_outproj(const float* __restrict__ x, const float* __restrict__ w,
                          const float* __restrict__ b, float* __restrict__ out,
                          int rows)
{
    int row  = blockIdx.x * 8 + (threadIdx.x >> 5);
    int lane = threadIdx.x & 31;
    if (row >= rows) return;
    float s = 0.0f;
    for (int c = lane; c < DMODEL; c += 32) s += x[(size_t)row * DMODEL + c] * w[c];
    for (int o = 16; o > 0; o >>= 1) s += __shfl_xor(s, o, 32);
    if (lane == 0) out[row] = s + b[0];
}

// ---------------------------------------------------------------------------
static inline void gemm(hipStream_t st, const float* A, const float* Bm,
                        long long bOff, int ldbK, int ldbN,
                        const float* bias, const float* Cin, float* Cout,
                        int M, int N, int K, int lda,
                        int ashift, int aseg, int flags)
{
    dim3 grid(N / 128, M / 128);
    k_gemm<<<grid, 256, 0, st>>>(A, Bm, bias, Cin, Cout, M, N, K, lda,
                                 bOff, ldbK, ldbN, ashift, aseg, flags);
}

extern "C" void kernel_launch(void* const* d_in, const int* in_sizes, int n_in,
                              void* d_out, int out_size, void* d_ws, size_t ws_size,
                              hipStream_t stream)
{
    (void)n_in; (void)out_size; (void)ws_size;
    const float* enc_x = (const float*)d_in[0];
    const float* dec_x = (const float*)d_in[1];
    #define P(i) ((const float*)d_in[(i)])
    const int* idx_enc0 = (const int*)d_in[63];
    const int* idx_enc1 = (const int*)d_in[64];
    const int* idx_dec  = (const int*)d_in[65];

    const int L0 = 2048, L1 = 1024, LD = 96;
    const int U0 = in_sizes[63] / L0;   // 38
    const int U1 = in_sizes[64] / L1;   // 34
    const int UD = in_sizes[65] / LD;   // 22
    const float scl = 0.125f;           // 1/sqrt(64)

    const size_t MB = 1ull << 20;
    char* ws = (char*)d_ws;
    float* X    = (float*)(ws);               // residual stream (16MB max)
    float* Qb   = (float*)(ws + 16 * MB);
    float* Kb   = (float*)(ws + 32 * MB);
    float* Vb   = (float*)(ws + 48 * MB);
    float* Ctx  = (float*)(ws + 64 * MB);
    float* FF   = (float*)(ws + 80 * MB);     // 64MB (ffn hidden)
    char*  sm   = ws + 144 * MB;
    float* Mbuf  = (float*)(sm);              // 256KB
    int*   Mtop  = (int*)(sm + 256 * 1024);
    float* vmean = (float*)(sm + 272 * 1024);
    float* Dd    = (float*)(ws + 145 * MB);   // decoder stream 384x512
    float* dQ    = (float*)(ws + 146 * MB);
    float* dK    = (float*)(ws + 147 * MB);
    float* dV    = (float*)(ws + 148 * MB);
    float* dCtx  = (float*)(ws + 149 * MB);
    float* dT    = (float*)(ws + 150 * MB);
    float* dFF   = (float*)(ws + 151 * MB);   // 3MB

    // ---------------- encoder embed ----------------
    {
        size_t tot = (size_t)NB * L0 * DMODEL;
        k_embed<<<(unsigned)(tot / 256), 256, 0, stream>>>(enc_x, P(2), P(3), X, L0, tot);
    }

    // ---------------- encoder layers ----------------
    int bases[2] = {6, 19};
    int Ls[2] = {L0, L1};
    int Us[2] = {U0, U1};
    const int* idxs[2] = {idx_enc0, idx_enc1};
    for (int li = 0; li < 2; ++li) {
        int p0 = bases[li];
        int L = Ls[li], u = Us[li];
        const int* idx = idxs[li];
        int rows = NB * L;
        size_t tot = (size_t)rows * DMODEL;

        // Q,K,V projections (no bias)
        gemm(stream, X, P(p0 + 0), 0, DMODEL, 1, 0, 0, Qb, rows, DMODEL, DMODEL, DMODEL, 0, 0, 0);
        gemm(stream, X, P(p0 + 1), 0, DMODEL, 1, 0, 0, Kb, rows, DMODEL, DMODEL, DMODEL, 0, 0, 0);
        gemm(stream, X, P(p0 + 2), 0, DMODEL, 1, 0, 0, Vb, rows, DMODEL, DMODEL, DMODEL, 0, 0, 0);
        // ProbSparse pieces
        k_sample<<<(NB * NHEAD * L) / 8, 256, 0, stream>>>(Qb, Kb, idx, Mbuf, L, u);
        k_topk<<<NB * NHEAD, 256, 0, stream>>>(Mbuf, Mtop, L, u);
        k_vmean<<<NB * NHEAD, 64, 0, stream>>>(Vb, vmean, L);
        k_fill<<<(unsigned)(tot / 256), 256, 0, stream>>>(Ctx, vmean, L, tot);
        k_attn<<<NB * NHEAD * u, 256, 0, stream>>>(Qb, Kb, Vb, Mtop, u, L, L, scl, Ctx);
        // O projection + residual LN
        gemm(stream, Ctx, P(p0 + 3), 0, DMODEL, 1, P(p0 + 4), 0, Qb, rows, DMODEL, DMODEL, DMODEL, 0, 0, 2);
        k_lnres<<<rows, 256, 0, stream>>>(X, Qb, P(p0 + 5), P(p0 + 6), X);
        // FFN
        gemm(stream, X, P(p0 + 7), 0, DFFN, 1, P(p0 + 8), 0, FF, rows, DFFN, DMODEL, DMODEL, 0, 0, 6);
        gemm(stream, FF, P(p0 + 9), 0, DMODEL, 1, P(p0 + 10), 0, Qb, rows, DMODEL, DFFN, DFFN, 0, 0, 2);
        k_lnres<<<rows, 256, 0, stream>>>(X, Qb, P(p0 + 11), P(p0 + 12), X);

        if (li == 0) {
            // distill: conv(k=3,p=1) as 3 shifted accumulating GEMMs into Qb
            // conv_w[o][i][t]: B(k=i,n=o) strides: ldbK=3, ldbN=1536, off=t
            for (int t = 0; t < 3; ++t)
                gemm(stream, X, P(32), t, 3, 1536, 0, Qb, Qb,
                     rows, DMODEL, DMODEL, DMODEL, t - 1, L, (t == 0) ? 0 : 1);
            k_bnelu<<<(unsigned)(tot / 256), 256, 0, stream>>>(
                Qb, P(33), P(34), P(35), P(36), P(37), tot);
            size_t tot1 = (size_t)NB * L1 * DMODEL;
            k_pool<<<(unsigned)(tot1 / 256), 256, 0, stream>>>(Qb, X, L0, L1, tot1);
        }
    }
    // encoder output e lives in X (NB*L1 rows)

    // ---------------- decoder ----------------
    {
        int rows = NB * LD;                     // 384
        size_t tot = (size_t)rows * DMODEL;
        k_embed<<<(unsigned)(tot / 256), 256, 0, stream>>>(dec_x, P(4), P(5), Dd, LD, tot);

        // self ProbSparse attention (params 38..42)
        gemm(stream, Dd, P(38), 0, DMODEL, 1, 0, 0, dQ, rows, DMODEL, DMODEL, DMODEL, 0, 0, 0);
        gemm(stream, Dd, P(39), 0, DMODEL, 1, 0, 0, dK, rows, DMODEL, DMODEL, DMODEL, 0, 0, 0);
        gemm(stream, Dd, P(40), 0, DMODEL, 1, 0, 0, dV, rows, DMODEL, DMODEL, DMODEL, 0, 0, 0);
        k_sample<<<(NB * NHEAD * LD) / 8, 256, 0, stream>>>(dQ, dK, idx_dec, Mbuf, LD, UD);
        k_topk<<<NB * NHEAD, 256, 0, stream>>>(Mbuf, Mtop, LD, UD);
        k_vmean<<<NB * NHEAD, 64, 0, stream>>>(dV, vmean, LD);
        k_fill<<<(unsigned)(tot / 256), 256, 0, stream>>>(dCtx, vmean, LD, tot);
        k_attn<<<NB * NHEAD * UD, 256, 0, stream>>>(dQ, dK, dV, Mtop, UD, LD, LD, scl, dCtx);
        gemm(stream, dCtx, P(41), 0, DMODEL, 1, P(42), 0, dT, rows, DMODEL, DMODEL, DMODEL, 0, 0, 2);
        k_lnres<<<rows, 256, 0, stream>>>(Dd, dT, P(51), P(52), Dd);

        // cross attention (params 43..50), K/V over encoder output (rows NB*L1)
        int erows = NB * L1;
        gemm(stream, Dd, P(43), 0, DMODEL, 1, P(44), 0, dQ, rows, DMODEL, DMODEL, DMODEL, 0, 0, 2);
        gemm(stream, X,  P(45), 0, DMODEL, 1, P(46), 0, Kb, erows, DMODEL, DMODEL, DMODEL, 0, 0, 2);
        gemm(stream, X,  P(47), 0, DMODEL, 1, P(48), 0, Vb, erows, DMODEL, DMODEL, DMODEL, 0, 0, 2);
        k_attn<<<NB * NHEAD * LD, 256, 0, stream>>>(dQ, Kb, Vb, (const int*)0, LD, LD, L1, scl, dCtx);
        gemm(stream, dCtx, P(49), 0, DMODEL, 1, P(50), 0, dT, rows, DMODEL, DMODEL, DMODEL, 0, 0, 2);
        k_lnres<<<rows, 256, 0, stream>>>(Dd, dT, P(53), P(54), Dd);

        // FFN (55..58) + LN3 (59,60)
        gemm(stream, Dd, P(55), 0, DFFN, 1, P(56), 0, dFF, rows, DFFN, DMODEL, DMODEL, 0, 0, 6);
        gemm(stream, dFF, P(57), 0, DMODEL, 1, P(58), 0, dT, rows, DMODEL, DFFN, DFFN, 0, 0, 2);
        k_lnres<<<rows, 256, 0, stream>>>(Dd, dT, P(59), P(60), Dd);

        // head (61,62)
        k_outproj<<<rows / 8, 256, 0, stream>>>(Dd, P(61), P(62), (float*)d_out, rows);
    }
    #undef P
}
